// BertForRelationExtractionV2_37735582663464
// MI455X (gfx1250) — compile-verified, matrix operand order checked by reading, not run on hardware
//
#include <hip/hip_runtime.h>
#include <hip/hip_bf16.h>
#include <math.h>

typedef __attribute__((ext_vector_type(16))) _Float16 v16h;
typedef __attribute__((ext_vector_type(8)))  float    v8f;

#define Hdim 768
#define SEQ  512
#define BATCH 32
#define NHS  8
#define HD   96          // head dim (768/8)
#define NROWS (BATCH*SEQ) // 16384

static __device__ __forceinline__ v8f wmma_f16(v16h a, v16h b, v8f c) {
    return __builtin_amdgcn_wmma_f32_16x16x32_f16(false, a, false, b, (short)0, c, false, false);
}

// ---------------- fp32 -> fp16 conversion ----------------
__global__ void cvt_f32_f16(const float* __restrict__ in, _Float16* __restrict__ out, int n) {
    int i = blockIdx.x * blockDim.x + threadIdx.x;
    if (i < n) out[i] = (_Float16)in[i];
}

// ---------------- WMMA GEMM: out = X(MxK) @ W(NxK)^T + bias ----------------
// mode 0: f16 out, layout [b][h][s][d]   (Q / K)
// mode 1: f16 out, layout [b][h][d][s]   (V transposed)
// mode 2: f32 out, row-major M x N       (O projection)
__global__ void gemm_xw(const _Float16* __restrict__ X, const _Float16* __restrict__ W,
                        const float* __restrict__ bias, void* __restrict__ out,
                        int M, int Kd, int N, int mode) {
    const int lane = threadIdx.x & 31;
    const int wave = threadIdx.x >> 5;
    const int rowTile = blockIdx.y * 8 + wave;          // 16 rows per tile
    const int colBase = blockIdx.x * 64;                // 4 x 16 cols per wave
    const int cL  = lane & 15;
    const int kHi = (lane >> 4) * 16;

    const _Float16* Arow = X + (size_t)(rowTile * 16 + cL) * Kd + kHi;
    const _Float16* Bcol = W + (size_t)(colBase + cL) * Kd + kHi;

    v8f acc0 = {}, acc1 = {}, acc2 = {}, acc3 = {};
    for (int k0 = 0; k0 < Kd; k0 += 32) {
        v16h a = *(const v16h*)(Arow + k0);
        __builtin_prefetch((const void*)(Arow + k0 + 128), 0, 1);
        v16h b0 = *(const v16h*)(Bcol + k0);
        v16h b1 = *(const v16h*)(Bcol + (size_t)16 * Kd + k0);
        v16h b2 = *(const v16h*)(Bcol + (size_t)32 * Kd + k0);
        v16h b3 = *(const v16h*)(Bcol + (size_t)48 * Kd + k0);
        acc0 = wmma_f16(a, b0, acc0);
        acc1 = wmma_f16(a, b1, acc1);
        acc2 = wmma_f16(a, b2, acc2);
        acc3 = wmma_f16(a, b3, acc3);
    }

    const int rOff = (lane >> 4) * 8;
    v8f accs[4] = {acc0, acc1, acc2, acc3};
    for (int t = 0; t < 4; ++t) {
        const int n = colBase + t * 16 + cL;
        const float bval = bias ? bias[n] : 0.0f;
        for (int i = 0; i < 8; ++i) {
            const int m = rowTile * 16 + i + rOff;
            const float v = accs[t][i] + bval;
            if (mode == 2) {
                ((float*)out)[(size_t)m * N + n] = v;
            } else {
                const int b = m >> 9, s = m & 511;
                const int h = n / HD, d = n % HD;
                if (mode == 0)
                    ((_Float16*)out)[(((size_t)b * NHS + h) * SEQ + s) * HD + d] = (_Float16)v;
                else
                    ((_Float16*)out)[(((size_t)b * NHS + h) * HD + d) * SEQ + s] = (_Float16)v;
            }
        }
    }
}

// ---------------- flash attention: one wave per (b, h, 16-query tile) ----------------
__global__ void attn(const _Float16* __restrict__ Q, const _Float16* __restrict__ K,
                     const _Float16* __restrict__ Vt, _Float16* __restrict__ Ctx) {
    __shared__ __align__(32) _Float16 Pld[16][32];   // P relayout buffer (C-layout -> A operand)
    const int lane = threadIdx.x;
    const int c    = lane & 15;
    const int half = lane >> 4;
    const int q0   = blockIdx.x * 16;
    const int h    = blockIdx.y;
    const int b    = blockIdx.z;

    const _Float16* Qh = Q  + ((size_t)(b * NHS + h) * SEQ) * HD;
    const _Float16* Kh = K  + ((size_t)(b * NHS + h) * SEQ) * HD;
    const _Float16* Vh = Vt + ((size_t)(b * NHS + h) * HD) * SEQ;

    // A operand for scores: Q tile (16 q x 96 d), loaded once
    v16h aQ0 = *(const v16h*)(Qh + (size_t)(q0 + c) * HD + 0  + half * 16);
    v16h aQ1 = *(const v16h*)(Qh + (size_t)(q0 + c) * HD + 32 + half * 16);
    v16h aQ2 = *(const v16h*)(Qh + (size_t)(q0 + c) * HD + 64 + half * 16);

    float m[8], l[8];
    v8f acc[6] = {};
    for (int i = 0; i < 8; ++i) { m[i] = -1e30f; l[i] = 0.0f; }
    const float scale = 0.10206207261596575f;  // 1/sqrt(96)

    for (int kb = 0; kb < SEQ / 32; ++kb) {
        const _Float16* Kb = Kh + (size_t)(kb * 32) * HD;
        v8f s0 = {}, s1 = {};
        {
            v16h b0 = *(const v16h*)(Kb + (size_t)c        * HD + 0  + half * 16);
            v16h b1 = *(const v16h*)(Kb + (size_t)(16 + c) * HD + 0  + half * 16);
            s0 = wmma_f16(aQ0, b0, s0);  s1 = wmma_f16(aQ0, b1, s1);
            b0 = *(const v16h*)(Kb + (size_t)c        * HD + 32 + half * 16);
            b1 = *(const v16h*)(Kb + (size_t)(16 + c) * HD + 32 + half * 16);
            s0 = wmma_f16(aQ1, b0, s0);  s1 = wmma_f16(aQ1, b1, s1);
            b0 = *(const v16h*)(Kb + (size_t)c        * HD + 64 + half * 16);
            b1 = *(const v16h*)(Kb + (size_t)(16 + c) * HD + 64 + half * 16);
            s0 = wmma_f16(aQ2, b0, s0);  s1 = wmma_f16(aQ2, b1, s1);
        }
        // online softmax update (rows i + 8*half; reduce across 16-lane halves)
        for (int i = 0; i < 8; ++i) {
            float x0 = s0[i] * scale, x1 = s1[i] * scale;
            float tm = fmaxf(x0, x1);
            tm = fmaxf(tm, __shfl_xor(tm, 1));
            tm = fmaxf(tm, __shfl_xor(tm, 2));
            tm = fmaxf(tm, __shfl_xor(tm, 4));
            tm = fmaxf(tm, __shfl_xor(tm, 8));
            float mn   = fmaxf(m[i], tm);
            float corr = __expf(m[i] - mn);
            float p0 = __expf(x0 - mn);
            float p1 = __expf(x1 - mn);
            float rs = p0 + p1;
            rs += __shfl_xor(rs, 1);
            rs += __shfl_xor(rs, 2);
            rs += __shfl_xor(rs, 4);
            rs += __shfl_xor(rs, 8);
            l[i] = l[i] * corr + rs;
            m[i] = mn;
            for (int t = 0; t < 6; ++t) acc[t][i] *= corr;
            // stash P (f16) in LDS in [q][k_local] order
            Pld[i + 8 * half][c]      = (_Float16)p0;
            Pld[i + 8 * half][16 + c] = (_Float16)p1;
        }
        // A operand for ctx: P tile (16 q x 32 k)
        v16h aP = *(const v16h*)(&Pld[c][half * 16]);
        for (int t = 0; t < 6; ++t) {
            v16h bv = *(const v16h*)(Vh + (size_t)(t * 16 + c) * SEQ + kb * 32 + half * 16);
            acc[t] = wmma_f16(aP, bv, acc[t]);
        }
    }

    // write ctx as row-major [b*S+s][h*96+d] f16 for the O-projection GEMM
    for (int t = 0; t < 6; ++t)
        for (int i = 0; i < 8; ++i) {
            const int r = i + 8 * half;
            const size_t mrow = (size_t)b * SEQ + q0 + r;
            const int col = h * HD + t * 16 + c;
            Ctx[mrow * Hdim + col] = (_Float16)(acc[t][i] / l[i]);
        }
}

// ---------------- entity pooling + positional gather ----------------
__global__ void pool(const float* __restrict__ Xo, const int* __restrict__ ep,
                     const float* __restrict__ pos_table,
                     float* __restrict__ e1, float* __restrict__ e2, float* __restrict__ pe) {
    const int b = blockIdx.x, t = threadIdx.x;
    const int p0 = ep[b * 4 + 0], p1 = ep[b * 4 + 1], p2 = ep[b * 4 + 2], p3 = ep[b * 4 + 3];
    const int s1 = max(p0 - 2, 0), e1e = min(p1 + 2, SEQ);
    const int s2 = max(p2 - 2, 0), e2e = min(p3 + 2, SEQ);
    const float inv1 = 1.0f / (float)(e1e - s1);
    const float inv2 = 1.0f / (float)(e2e - s2);
    for (int c = t; c < Hdim; c += 256) {
        float a1 = 0.0f, a2 = 0.0f;
        for (int s = s1; s < e1e; ++s) a1 += Xo[((size_t)b * SEQ + s) * Hdim + c];
        for (int s = s2; s < e2e; ++s) a2 += Xo[((size_t)b * SEQ + s) * Hdim + c];
        e1[b * Hdim + c] = a1 * inv1;
        e2[b * Hdim + c] = a2 * inv2;
        pe[b * Hdim + c] = pos_table[(size_t)p0 * Hdim + c] + pos_table[(size_t)p1 * Hdim + c]
                         + pos_table[(size_t)p2 * Hdim + c] + pos_table[(size_t)p3 * Hdim + c];
    }
}

// ---------------- cross-attn (softmax over size-1 axis == identity) + LN + classifier ----------------
__global__ void head(const float* __restrict__ e1g, const float* __restrict__ e2g,
                     const float* __restrict__ peg,
                     const float* __restrict__ ciw, const float* __restrict__ cib,
                     const float* __restrict__ cow, const float* __restrict__ cob,
                     const float* __restrict__ lng, const float* __restrict__ lnb,
                     const float* __restrict__ clw, const float* __restrict__ clb,
                     float* __restrict__ out) {
    __shared__ float e1s[Hdim], e2s[Hdim], vhs[Hdim], comb[6 * Hdim], red[256], red2[256];
    const int b = blockIdx.x, t = threadIdx.x;
    for (int c = t; c < Hdim; c += 256) { e1s[c] = e1g[b * Hdim + c]; e2s[c] = e2g[b * Hdim + c]; }
    __syncthreads();
    // vh = e2 @ vw^T + vb   (vw = cross_in_w rows [2H, 3H))
    for (int c = t; c < Hdim; c += 256) {
        float a = cib[2 * Hdim + c];
        const float* wr = ciw + (size_t)(2 * Hdim + c) * Hdim;
        for (int k = 0; k < Hdim; ++k) a += e2s[k] * wr[k];
        vhs[c] = a;
    }
    __syncthreads();
    for (int c = t; c < Hdim; c += 256) {
        float a = cob[c];
        const float* wr = cow + (size_t)c * Hdim;
        for (int k = 0; k < Hdim; ++k) a += vhs[k] * wr[k];
        const float v1 = e1s[c], v2 = e2s[c];
        comb[c] = v1;  comb[Hdim + c] = v2;  comb[2 * Hdim + c] = v1 * v2;
        comb[3 * Hdim + c] = v1 - v2;  comb[4 * Hdim + c] = peg[b * Hdim + c];
        comb[5 * Hdim + c] = a;
    }
    __syncthreads();
    float s = 0.0f, sq = 0.0f;
    for (int i = t; i < 6 * Hdim; i += 256) { const float x = comb[i]; s += x; sq += x * x; }
    red[t] = s; red2[t] = sq; __syncthreads();
    for (int o = 128; o > 0; o >>= 1) {
        if (t < o) { red[t] += red[t + o]; red2[t] += red2[t + o]; }
        __syncthreads();
    }
    const float mu  = red[0] / (6.0f * Hdim);
    const float var = red2[0] / (6.0f * Hdim) - mu * mu;
    const float rstd = rsqrtf(var + 1e-5f);
    if (t < 14) {
        float a = clb[t];
        const float* wr = clw + (size_t)t * (6 * Hdim);
        for (int i = 0; i < 6 * Hdim; ++i) {
            const float f = (comb[i] - mu) * rstd * lng[i] + lnb[i];
            a += f * wr[i];
        }
        out[b * 14 + t] = a;
    }
}

extern "C" void kernel_launch(void* const* d_in, const int* in_sizes, int n_in,
                              void* d_out, int out_size, void* d_ws, size_t ws_size,
                              hipStream_t stream) {
    const float* seq_out = (const float*)d_in[0];
    const int*   ep      = (const int*)d_in[1];
    const float* Wq = (const float*)d_in[2];  const float* bq = (const float*)d_in[3];
    const float* Wk = (const float*)d_in[4];  const float* bk = (const float*)d_in[5];
    const float* Wv = (const float*)d_in[6];  const float* bv = (const float*)d_in[7];
    const float* Wo = (const float*)d_in[8];  const float* bo = (const float*)d_in[9];
    const float* pos_table = (const float*)d_in[10];
    const float* ciw = (const float*)d_in[11]; const float* cib = (const float*)d_in[12];
    const float* cow = (const float*)d_in[13]; const float* cob = (const float*)d_in[14];
    const float* lng = (const float*)d_in[15]; const float* lnb = (const float*)d_in[16];
    const float* clw = (const float*)d_in[17]; const float* clb = (const float*)d_in[18];
    float* out = (float*)d_out;

    // workspace carve-out (all sizes 256B-aligned)
    char* ws = (char*)d_ws;
    const size_t XH  = (size_t)NROWS * Hdim * 2;   // 24 MB f16 activation
    const size_t WB  = (size_t)Hdim * Hdim * 2;    // f16 weight
    const size_t HB  = XH;                         // per-head f16 tensors
    _Float16* Xh  = (_Float16*)(ws);
    _Float16* Wqh = (_Float16*)(ws + XH);
    _Float16* Wkh = (_Float16*)(ws + XH + WB);
    _Float16* Wvh = (_Float16*)(ws + XH + 2 * WB);
    _Float16* Woh = (_Float16*)(ws + XH + 3 * WB);
    _Float16* Qh  = (_Float16*)(ws + XH + 4 * WB);
    _Float16* Kh  = (_Float16*)(ws + XH + 4 * WB + HB);
    _Float16* Vt  = (_Float16*)(ws + XH + 4 * WB + 2 * HB);
    _Float16* Ctx = (_Float16*)(ws + XH + 4 * WB + 3 * HB);
    float*    Xo  = (float*)   (ws + XH + 4 * WB + 4 * HB);
    float*    e1  = (float*)   (ws + XH + 4 * WB + 4 * HB + (size_t)NROWS * Hdim * 4);
    float*    e2  = e1 + BATCH * Hdim;
    float*    pe  = e2 + BATCH * Hdim;

    // 1. convert activation + weights to f16
    {
        int n = NROWS * Hdim;
        cvt_f32_f16<<<(n + 255) / 256, 256, 0, stream>>>(seq_out, Xh, n);
        n = Hdim * Hdim;
        cvt_f32_f16<<<(n + 255) / 256, 256, 0, stream>>>(Wq, Wqh, n);
        cvt_f32_f16<<<(n + 255) / 256, 256, 0, stream>>>(Wk, Wkh, n);
        cvt_f32_f16<<<(n + 255) / 256, 256, 0, stream>>>(Wv, Wvh, n);
        cvt_f32_f16<<<(n + 255) / 256, 256, 0, stream>>>(Wo, Woh, n);
    }
    // 2. QKV projections (WMMA)
    {
        dim3 grid(Hdim / 64, NROWS / 128);
        gemm_xw<<<grid, 256, 0, stream>>>(Xh, Wqh, bq, Qh, NROWS, Hdim, Hdim, 0);
        gemm_xw<<<grid, 256, 0, stream>>>(Xh, Wkh, bk, Kh, NROWS, Hdim, Hdim, 0);
        gemm_xw<<<grid, 256, 0, stream>>>(Xh, Wvh, bv, Vt, NROWS, Hdim, Hdim, 1);
    }
    // 3. flash attention (WMMA)
    attn<<<dim3(SEQ / 16, NHS, BATCH), 32, 0, stream>>>(Qh, Kh, Vt, Ctx);
    // 4. output projection (WMMA, fp32 out)
    gemm_xw<<<dim3(Hdim / 64, NROWS / 128), 256, 0, stream>>>(Ctx, Woh, bo, Xo, NROWS, Hdim, Hdim, 2);
    // 5. entity pooling + pos-emb
    pool<<<BATCH, 256, 0, stream>>>(Xo, ep, pos_table, e1, e2, pe);
    // 6. cross + layernorm + classifier
    head<<<BATCH, 256, 0, stream>>>(e1, e2, pe, ciw, cib, cow, cob, lng, lnb, clw, clb, out);
}
